// HierarchicalStage_26190710571581
// MI455X (gfx1250) — matrix operand order repaired and checked by reading, not compiled, not run on hardware
//
#include <hip/hip_runtime.h>
#include <cstddef>

// ---------------- CDNA5 WMMA types ----------------
typedef __attribute__((ext_vector_type(16))) _Float16 v16h;
typedef __attribute__((ext_vector_type(8)))  float    v8f;

// Model constants
#define BB   16
#define NN   1024
#define TT   1025
#define TP   1040      // padded token count (65 * 16)
#define EE   384
#define HH   6
#define DH   64
#define FFD  1536
#define KSEL 256

// ---------------- gfx1250 async global->LDS copy (guarded; falls back cleanly) ----------------
#if defined(__has_builtin)
# if __has_builtin(__builtin_amdgcn_global_load_async_to_lds_b128)
#  define ASYNC_OK 1
# endif
#endif

#ifdef ASYNC_OK
// builtin signature: (int4 addrspace(1)*, int4 addrspace(3)*, imm offset, imm cpol)
typedef int v4i __attribute__((vector_size(16)));
typedef __attribute__((address_space(1))) v4i* gv4i_p;
typedef __attribute__((address_space(3))) v4i* lv4i_p;

__device__ inline void async_cp16(const _Float16* g, _Float16* l) {
  __builtin_amdgcn_global_load_async_to_lds_b128((gv4i_p)g, (lv4i_p)l, 0, 0);
}
__device__ inline void wait_async0() {
# if __has_builtin(__builtin_amdgcn_s_wait_asynccnt)
  __builtin_amdgcn_s_wait_asynccnt(0);
# else
  asm volatile("s_wait_asynccnt 0" ::: "memory");
# endif
}
#endif

// ---------------- fragment loaders (contiguous k = 16*hi + i, wave32), f16 LDS ----------------
// A-matrix 16x32: lane l -> row m = l&15, half hi = l>>4. Pairs contiguous -> 2x ds_load_b128.
__device__ inline v16h load_a_frag(const _Float16* __restrict__ tile, int ld, int lane) {
  int m = lane & 15, hi = lane >> 4;
  const _Float16* row = tile + (size_t)m * ld;
  v16h a;
#pragma unroll
  for (int i = 0; i < 16; ++i) {
    int v = i >> 1, p = i & 1;
    int k = hi * 16 + v * 2 + p;
    a[i] = row[k];
  }
  return a;
}

// B-matrix 32x16, tile stored n-major: element (k,n) at tile[n*ld + k]. 2x ds_load_b128.
__device__ inline v16h load_b_frag(const _Float16* __restrict__ tile, int ld, int lane) {
  int n = lane & 15, hi = lane >> 4;
  const _Float16* row = tile + (size_t)n * ld + hi * 16;
  v16h b;
#pragma unroll
  for (int i = 0; i < 16; ++i) b[i] = row[i];
  return b;
}

// B-matrix 32x16, tile stored k-major: element (k,n) at tile[k*ld + n].
__device__ inline v16h load_b_frag_kmajor(const _Float16* __restrict__ tile, int ld, int lane) {
  int n = lane & 15, hi = lane >> 4;
  v16h b;
#pragma unroll
  for (int i = 0; i < 16; ++i) b[i] = tile[(size_t)(hi * 16 + i) * ld + n];
  return b;
}

__device__ inline v8f wmma16(v16h a, v16h b, v8f c) {
  return __builtin_amdgcn_wmma_f32_16x16x32_f16(false, a, false, b, (short)0, c, false, false);
}

// ---------------- f32 -> f16 bulk convert (n multiple of 4) ----------------
__global__ void cvt_f16_kernel(const float* __restrict__ in, _Float16* __restrict__ out, int n) {
  int i = (blockIdx.x * 256 + threadIdx.x) * 4;
  if (i + 3 < n) {
    float4 v = *(const float4*)(in + i);
    _Float16* o = out + i;
    o[0] = (_Float16)v.x; o[1] = (_Float16)v.y; o[2] = (_Float16)v.z; o[3] = (_Float16)v.w;
  }
}

// ---------------- GEMM: C[M,N] = A[M,K](f16) * W[N,K](f16)^T + bias (+R) (ReLU?) ------------
// async double-buffered LDS pipeline; f32 out (nullable) + f16 mirror out (nullable)
#define BM 128
#define BN 64
#define BK 32
#define GEMM_THREADS 256   // 8 waves

__global__ void gemm_kernel(const _Float16* __restrict__ A, const _Float16* __restrict__ W,
                            const float* __restrict__ bias, const float* __restrict__ R,
                            float* __restrict__ C, _Float16* __restrict__ Ch,
                            int M, int Nn, int Kk, int relu) {
  __shared__ __align__(16) _Float16 As[2][BM][BK];   // 2 x 8KB
  __shared__ __align__(16) _Float16 Bs[2][BN][BK];   // 2 x 4KB
  int tid  = threadIdx.x;
  int lane = tid & 31;
  int wave = tid >> 5;               // 0..7, 16 rows each
  int m0 = blockIdx.y * BM;
  int n0 = blockIdx.x * BN;
  bool full = (m0 + BM <= M) && (n0 + BN <= Nn);  // K always multiple of BK here

  // stage one K-tile into buffer `buf` (16B granules = 8 halves)
  auto stage = [&](int k0, int buf) {
    if (full) {
#ifdef ASYNC_OK
#pragma unroll
      for (int p = 0; p < 2; ++p) {
        int g = tid + p * GEMM_THREADS;   // 0..511  (A: 128 rows x 4 granules)
        int r = g >> 2, c8 = g & 3;
        async_cp16(&A[(size_t)(m0 + r) * Kk + k0 + c8 * 8], &As[buf][r][c8 * 8]);
      }
      {
        int g = tid;                       // 0..255  (B: 64 rows x 4 granules)
        int r = g >> 2, c8 = g & 3;
        async_cp16(&W[(size_t)(n0 + r) * Kk + k0 + c8 * 8], &Bs[buf][r][c8 * 8]);
      }
#else
#pragma unroll
      for (int p = 0; p < 2; ++p) {
        int g = tid + p * GEMM_THREADS;
        int r = g >> 2, c8 = g & 3;
        *(uint4*)&As[buf][r][c8 * 8] = *(const uint4*)&A[(size_t)(m0 + r) * Kk + k0 + c8 * 8];
      }
      {
        int r = tid >> 2, c8 = tid & 3;
        *(uint4*)&Bs[buf][r][c8 * 8] = *(const uint4*)&W[(size_t)(n0 + r) * Kk + k0 + c8 * 8];
      }
#endif
    } else {
      uint4 z = make_uint4(0, 0, 0, 0);
#pragma unroll
      for (int p = 0; p < 2; ++p) {
        int g = tid + p * GEMM_THREADS;
        int r = g >> 2, c8 = g & 3;
        uint4 t = (m0 + r < M) ? *(const uint4*)&A[(size_t)(m0 + r) * Kk + k0 + c8 * 8] : z;
        *(uint4*)&As[buf][r][c8 * 8] = t;
      }
      {
        int r = tid >> 2, c8 = tid & 3;
        uint4 t = (n0 + r < Nn) ? *(const uint4*)&W[(size_t)(n0 + r) * Kk + k0 + c8 * 8] : z;
        *(uint4*)&Bs[buf][r][c8 * 8] = t;
      }
    }
  };

  v8f acc[4] = {};

  stage(0, 0);
#ifdef ASYNC_OK
  if (full) wait_async0();
#endif
  __syncthreads();

  int cur = 0;
  for (int k0 = 0; k0 < Kk; k0 += BK) {
    bool more = (k0 + BK < Kk);
    if (more) stage(k0 + BK, 1 - cur);    // overlap next-tile copy with WMMAs below

    v16h a = load_a_frag(&As[cur][wave * 16][0], BK, lane);
#pragma unroll
    for (int nt = 0; nt < 4; ++nt) {
      v16h b = load_b_frag(&Bs[cur][nt * 16][0], BK, lane);
      acc[nt] = wmma16(a, b, acc[nt]);
    }

#ifdef ASYNC_OK
    if (more && full) wait_async0();
#endif
    __syncthreads();
    cur ^= 1;
  }

  // epilogue: c[r] <-> (m = r + 8*hi, n = lane&15)
  int hi = lane >> 4, ln = lane & 15;
#pragma unroll
  for (int nt = 0; nt < 4; ++nt) {
    int gn = n0 + nt * 16 + ln;
    if (gn >= Nn) continue;
    float bb = bias ? bias[gn] : 0.f;
#pragma unroll
    for (int r = 0; r < 8; ++r) {
      int gm = m0 + wave * 16 + hi * 8 + r;
      if (gm >= M) continue;
      float v = acc[nt][r] + bb;
      if (R) v += R[(size_t)gm * Nn + gn];
      if (relu) v = v > 0.f ? v : 0.f;
      size_t idx = (size_t)gm * Nn + gn;
      if (C)  C[idx]  = v;
      if (Ch) Ch[idx] = (_Float16)v;
    }
  }
}

// ---------------- assemble x = [cls ; tokens + PE + pos], zero pad rows ----------------
__global__ void assemble_kernel(const float* __restrict__ tok, const float* __restrict__ coords,
                                const float* __restrict__ cls, const float* __restrict__ pos,
                                float* __restrict__ X, _Float16* __restrict__ Xh) {
  int t = blockIdx.x, b = blockIdx.y, e = threadIdx.x;  // block = 384
  float v;
  if (t == 0) {
    v = cls[e] + pos[e];
  } else if (t <= NN) {
    int n = t - 1;
    float px = coords[((size_t)b * NN + n) * 2 + 0];
    float py = coords[((size_t)b * NN + n) * 2 + 1];
    int q = e / 96, j = e % 96;
    float dv  = __expf(-0.0479705222f * (float)(2 * j));  // ln(10000)/192
    float arg = ((q < 2) ? px : py) * dv;
    float pe  = (q & 1) ? __cosf(arg) : __sinf(arg);
    v = tok[((size_t)b * NN + n) * EE + e] + pe + pos[(size_t)t * EE + e];
  } else {
    v = 0.f;
  }
  size_t idx = ((size_t)b * TP + t) * EE + e;
  X[idx]  = v;
  Xh[idx] = (_Float16)v;
}

// ---------------- flash attention: 1 wave per (b, h, 16-row q tile); f16 in/out -------------
__global__ void attn_kernel(const _Float16* __restrict__ QKV, _Float16* __restrict__ O) {
  __shared__ __align__(16) _Float16 Qs[16][64];
  __shared__ __align__(16) _Float16 Ks[32][64];
  __shared__ __align__(16) _Float16 Vs[32][64];
  __shared__ __align__(16) _Float16 Ps[16][32];

  int qt = blockIdx.x, h = blockIdx.y, b = blockIdx.z;
  int lane = threadIdx.x;
  int hi = lane >> 4, ln = lane & 15;
  int q0 = qt * 16;
  const size_t rs = 3 * EE;  // 1152 (halves)
  const _Float16* base = QKV + (size_t)b * TP * rs;

  // stage Q tile: row = lane>>1, half-row (4 x b128) = lane&1
  {
    int r = lane >> 1, hf = lane & 1;
    int t = q0 + r;
    if (t < TT) {
      const _Float16* qr = base + (size_t)t * rs + h * DH + hf * 32;
#ifdef ASYNC_OK
#pragma unroll
      for (int c8 = 0; c8 < 4; ++c8) async_cp16(qr + c8 * 8, &Qs[r][hf * 32 + c8 * 8]);
#else
#pragma unroll
      for (int c8 = 0; c8 < 4; ++c8)
        *(uint4*)&Qs[r][hf * 32 + c8 * 8] = *(const uint4*)(qr + c8 * 8);
#endif
    } else {
      uint4 z = make_uint4(0, 0, 0, 0);
#pragma unroll
      for (int c8 = 0; c8 < 4; ++c8) *(uint4*)&Qs[r][hf * 32 + c8 * 8] = z;
    }
#ifdef ASYNC_OK
    wait_async0();
#endif
  }
  __syncthreads();

  float rowM[8], rowL[8];
#pragma unroll
  for (int r = 0; r < 8; ++r) { rowM[r] = -1e30f; rowL[r] = 0.f; }
  v8f o0 = {}, o1 = {}, o2 = {}, o3 = {};

  for (int kt0 = 0; kt0 < TT; kt0 += 32) {
    // stage K/V tiles: one 64-half row (8 x b128) per lane
    {
      int t = kt0 + lane;
      if (t < TT) {
        const _Float16* kr = base + (size_t)t * rs + EE + h * DH;
        const _Float16* vr = base + (size_t)t * rs + 2 * EE + h * DH;
#ifdef ASYNC_OK
#pragma unroll
        for (int c8 = 0; c8 < 8; ++c8) {
          async_cp16(kr + c8 * 8, &Ks[lane][c8 * 8]);
          async_cp16(vr + c8 * 8, &Vs[lane][c8 * 8]);
        }
#else
#pragma unroll
        for (int c8 = 0; c8 < 8; ++c8) {
          *(uint4*)&Ks[lane][c8 * 8] = *(const uint4*)(kr + c8 * 8);
          *(uint4*)&Vs[lane][c8 * 8] = *(const uint4*)(vr + c8 * 8);
        }
#endif
      } else {
        uint4 z = make_uint4(0, 0, 0, 0);
#pragma unroll
        for (int c8 = 0; c8 < 8; ++c8) {
          *(uint4*)&Ks[lane][c8 * 8] = z;
          *(uint4*)&Vs[lane][c8 * 8] = z;
        }
      }
#ifdef ASYNC_OK
      wait_async0();
#endif
    }
    __syncthreads();

    v8f s0 = {}, s1 = {};
#pragma unroll
    for (int dc = 0; dc < 2; ++dc) {
      v16h aq  = load_a_frag(&Qs[0][dc * 32], 64, lane);
      v16h bk0 = load_b_frag(&Ks[0][dc * 32], 64, lane);
      v16h bk1 = load_b_frag(&Ks[16][dc * 32], 64, lane);
      s0 = wmma16(aq, bk0, s0);
      s1 = wmma16(aq, bk1, s1);
    }

    const float scale = 0.125f;  // 1/sqrt(64)
    float mx[8];
#pragma unroll
    for (int r = 0; r < 8; ++r) {
      float a0 = s0[r] * scale, a1 = s1[r] * scale;
      if (kt0 + ln >= TT)      a0 = -1e30f;
      if (kt0 + 16 + ln >= TT) a1 = -1e30f;
      s0[r] = a0; s1[r] = a1;
      mx[r] = fmaxf(a0, a1);
    }
#pragma unroll
    for (int off = 1; off < 16; off <<= 1)
#pragma unroll
      for (int r = 0; r < 8; ++r) mx[r] = fmaxf(mx[r], __shfl_xor(mx[r], off, 32));

    float rsum[8];
#pragma unroll
    for (int r = 0; r < 8; ++r) {
      float newM  = fmaxf(rowM[r], mx[r]);
      float alpha = __expf(rowM[r] - newM);
      rowM[r] = newM;
      float p0 = __expf(s0[r] - newM);
      float p1 = __expf(s1[r] - newM);
      s0[r] = p0; s1[r] = p1;
      rsum[r] = p0 + p1;
      o0[r] *= alpha; o1[r] *= alpha; o2[r] *= alpha; o3[r] *= alpha;
      rowL[r] *= alpha;
    }
#pragma unroll
    for (int off = 1; off < 16; off <<= 1)
#pragma unroll
      for (int r = 0; r < 8; ++r) rsum[r] += __shfl_xor(rsum[r], off, 32);
#pragma unroll
    for (int r = 0; r < 8; ++r) rowL[r] += rsum[r];

    __syncthreads();
#pragma unroll
    for (int r = 0; r < 8; ++r) {
      Ps[8 * hi + r][ln]      = (_Float16)s0[r];
      Ps[8 * hi + r][16 + ln] = (_Float16)s1[r];
    }
    __syncthreads();

    v16h ap = load_a_frag(&Ps[0][0], 32, lane);
    o0 = wmma16(ap, load_b_frag_kmajor(&Vs[0][0],  64, lane), o0);
    o1 = wmma16(ap, load_b_frag_kmajor(&Vs[0][16], 64, lane), o1);
    o2 = wmma16(ap, load_b_frag_kmajor(&Vs[0][32], 64, lane), o2);
    o3 = wmma16(ap, load_b_frag_kmajor(&Vs[0][48], 64, lane), o3);
    __syncthreads();
  }

#pragma unroll
  for (int r = 0; r < 8; ++r) {
    int m = 8 * hi + r;
    int t = q0 + m;
    if (t >= TT) continue;
    float inv = 1.f / (rowL[r] + 1e-20f);
    size_t out = ((size_t)b * TP + t) * EE + h * DH;
    O[out + 0 * 16 + ln] = (_Float16)(o0[r] * inv);
    O[out + 1 * 16 + ln] = (_Float16)(o1[r] * inv);
    O[out + 2 * 16 + ln] = (_Float16)(o2[r] * inv);
    O[out + 3 * 16 + ln] = (_Float16)(o3[r] * inv);
  }
}

// ---------------- row LayerNorm over E=384 (strided input rows, f32 + f16 mirror out) --------
__global__ void layernorm_kernel(const float* __restrict__ In, long long rowStride,
                                 const float* __restrict__ w, const float* __restrict__ bsh,
                                 float* __restrict__ Out, _Float16* __restrict__ Outh) {
  __shared__ float sm[8];
  int row = blockIdx.x, tid = threadIdx.x;  // block = 128
  const float* x = In + (size_t)row * rowStride;
  float v0 = x[tid], v1 = x[tid + 128], v2 = x[tid + 256];
  float s = v0 + v1 + v2;
  float s2 = v0 * v0 + v1 * v1 + v2 * v2;
#pragma unroll
  for (int off = 16; off; off >>= 1) {
    s  += __shfl_xor(s,  off, 32);
    s2 += __shfl_xor(s2, off, 32);
  }
  int wv = tid >> 5, ln = tid & 31;
  if (ln == 0) { sm[wv] = s; sm[4 + wv] = s2; }
  __syncthreads();
  if (tid == 0) {
    float ts = sm[0] + sm[1] + sm[2] + sm[3];
    float t2 = sm[4] + sm[5] + sm[6] + sm[7];
    float mean = ts / 384.f;
    float var  = t2 / 384.f - mean * mean;
    sm[0] = mean;
    sm[1] = rsqrtf(var + 1e-5f);
  }
  __syncthreads();
  float mean = sm[0], inv = sm[1];
  size_t rb = (size_t)row * EE;
#pragma unroll
  for (int j = 0; j < 3; ++j) {
    int e = tid + j * 128;
    float vv = ((j == 0 ? v0 : (j == 1 ? v1 : v2)) - mean) * inv * w[e] + bsh[e];
    Out[rb + e] = vv;
    if (Outh) Outh[rb + e] = (_Float16)vv;
  }
}

// ---------------- top-K via in-LDS bitonic sort (descending), K=256 of 1024 ----------------
__global__ void topk_kernel(const float* __restrict__ logits, const float* __restrict__ gumbel,
                            int* __restrict__ idx_int, float* __restrict__ idx_out) {
  __shared__ float key[1024];
  __shared__ int   id[1024];
  int b = blockIdx.x, t = threadIdx.x;
  key[t] = logits[b * 1024 + t] + gumbel[b * 1024 + t];  // softmax monotone: order on pre-activation
  id[t] = t;
  __syncthreads();
  for (int ksz = 2; ksz <= 1024; ksz <<= 1) {
    for (int j = ksz >> 1; j > 0; j >>= 1) {
      int p = t ^ j;
      if (p > t) {
        bool desc = ((t & ksz) == 0);
        float ka = key[t], kb = key[p];
        bool sw = desc ? (ka < kb) : (ka > kb);
        if (sw) {
          int ia = id[t], ib = id[p];
          key[t] = kb; key[p] = ka;
          id[t] = ib;  id[p] = ia;
        }
      }
      __syncthreads();
    }
  }
  if (t < KSEL) {
    idx_int[b * KSEL + t] = id[t];
    idx_out[b * KSEL + t] = (float)id[t];
  }
}

// ---------------- gather selected patches -> 4x (8x8) sub-patches + coords ----------------
__global__ void gather_kernel(const float* __restrict__ raw, const float* __restrict__ coords,
                              const int* __restrict__ idxbuf,
                              float* __restrict__ pout, float* __restrict__ cout) {
  int k = blockIdx.x, b = blockIdx.y, tid = threadIdx.x;  // block = 256
  int pidx = idxbuf[b * KSEL + k];
  const float* src = raw + ((size_t)b * NN + pidx) * 768;
  for (int u = tid; u < 768; u += 256) {
    int s = u / 192, f = u % 192;
    int c = f / 64, py = (f & 63) >> 3, px = f & 7;
    int sy = s >> 1, sx = s & 1;
    float v = src[c * 256 + (sy * 8 + py) * 16 + (sx * 8 + px)];
    pout[((size_t)b * 1024 + k * 4 + s) * 192 + f] = v;
  }
  if (tid < 8) {
    int s = tid >> 1, comp = tid & 1;
    float base = coords[((size_t)b * NN + pidx) * 2 + comp];
    float add  = (comp == 0) ? 8.f * (float)(s & 1) : 8.f * (float)(s >> 1);
    cout[((size_t)b * 1024 + k * 4 + s) * 2 + comp] = base + add;
  }
}

// ---------------- host-side launch ----------------
static inline dim3 cvt_grid(int n) { return dim3((n / 4 + 255) / 256); }

extern "C" void kernel_launch(void* const* d_in, const int* in_sizes, int n_in,
                              void* d_out, int out_size, void* d_ws, size_t ws_size,
                              hipStream_t stream) {
  (void)in_sizes; (void)n_in; (void)out_size; (void)ws_size;
  const float* raw     = (const float*)d_in[0];
  const float* coords  = (const float*)d_in[1];
  const float* W_embed = (const float*)d_in[2];
  const float* b_embed = (const float*)d_in[3];
  const float* cls_tok = (const float*)d_in[4];
  const float* pos     = (const float*)d_in[5];
  const float* Wqkv    = (const float*)d_in[6];
  const float* bqkv    = (const float*)d_in[7];
  const float* Wo      = (const float*)d_in[8];
  const float* bo      = (const float*)d_in[9];
  const float* ln1w    = (const float*)d_in[10];
  const float* ln1b    = (const float*)d_in[11];
  const float* W1      = (const float*)d_in[12];
  const float* b1      = (const float*)d_in[13];
  const float* W2      = (const float*)d_in[14];
  const float* b2      = (const float*)d_in[15];
  const float* ln2w    = (const float*)d_in[16];
  const float* ln2b    = (const float*)d_in[17];
  const float* lnfw    = (const float*)d_in[18];
  const float* lnfb    = (const float*)d_in[19];
  const float* Wsc     = (const float*)d_in[20];
  const float* bsc     = (const float*)d_in[21];
  const float* gumbel  = (const float*)d_in[22];

  // -------- f32 workspace --------
  float* ws = (float*)d_ws;
  size_t off = 0;
  float* X      = ws + off; off += (size_t)BB * TP * EE;   // 6.39M
  float* Y      = ws + off; off += (size_t)BB * TP * EE;   // tokens / residual out
  float* clsb   = ws + off; off += (size_t)BB * EE;
  float* logits = ws + off; off += (size_t)BB * 1024;
  int*   idxint = (int*)(ws + off); off += (size_t)BB * KSEL;

  // -------- f16 mirrors (after f32 region, 16B aligned) --------
  _Float16* hb = (_Float16*)(ws + off);
  size_t ho = 0;
  _Float16* rawh  = hb + ho; ho += (size_t)BB * NN * 768;       // 12.58M
  _Float16* weh   = hb + ho; ho += (size_t)EE * 768;
  _Float16* wqkvh = hb + ho; ho += (size_t)2 * 1152 * EE;
  _Float16* woh   = hb + ho; ho += (size_t)2 * EE * EE;
  _Float16* w1h   = hb + ho; ho += (size_t)2 * FFD * EE;
  _Float16* w2h   = hb + ho; ho += (size_t)2 * EE * FFD;
  _Float16* wsch  = hb + ho; ho += (size_t)1024 * EE;
  _Float16* Xh    = hb + ho; ho += (size_t)BB * TP * EE;
  _Float16* Obh   = hb + ho; ho += (size_t)BB * TP * EE;
  _Float16* Hbh   = hb + ho; ho += (size_t)BB * TP * FFD;       // qkv (1152) / ffn hidden (1536)
  _Float16* clsh  = hb + ho; ho += (size_t)BB * EE;

  float* pout = (float*)d_out;
  float* cout = pout + (size_t)BB * 1024 * 192;
  float* iout = cout + (size_t)BB * 1024 * 2;

  const int Mrows = BB * TP;  // 16640
  dim3 blk(GEMM_THREADS);

  // one-time f32 -> f16 conversions (inputs + weights)
  cvt_f16_kernel<<<cvt_grid(BB * NN * 768), 256, 0, stream>>>(raw, rawh, BB * NN * 768);
  cvt_f16_kernel<<<cvt_grid(EE * 768), 256, 0, stream>>>(W_embed, weh, EE * 768);
  cvt_f16_kernel<<<cvt_grid(2 * 1152 * EE), 256, 0, stream>>>(Wqkv, wqkvh, 2 * 1152 * EE);
  cvt_f16_kernel<<<cvt_grid(2 * EE * EE), 256, 0, stream>>>(Wo, woh, 2 * EE * EE);
  cvt_f16_kernel<<<cvt_grid(2 * FFD * EE), 256, 0, stream>>>(W1, w1h, 2 * FFD * EE);
  cvt_f16_kernel<<<cvt_grid(2 * EE * FFD), 256, 0, stream>>>(W2, w2h, 2 * EE * FFD);
  cvt_f16_kernel<<<cvt_grid(1024 * EE), 256, 0, stream>>>(Wsc, wsch, 1024 * EE);

  // patch embed: tokens = raw @ W_embed^T + b_embed   (M=16384, N=384, K=768)
  gemm_kernel<<<dim3(6, 128), blk, 0, stream>>>(rawh, weh, b_embed, nullptr, Y, nullptr,
                                                BB * NN, EE, 768, 0);
  // x = [cls ; tokens + sincos(coords) + pos], pad rows zeroed
  assemble_kernel<<<dim3(TP, BB), dim3(EE), 0, stream>>>(Y, coords, cls_tok, pos, X, Xh);

  for (int i = 0; i < 2; ++i) {
    // qkv = x @ Wqkv^T + bqkv   (N=1152) -> f16 only
    gemm_kernel<<<dim3(18, 130), blk, 0, stream>>>(Xh, wqkvh + (size_t)i * 1152 * EE,
                                                   bqkv + i * 1152, nullptr, nullptr, Hbh,
                                                   Mrows, 1152, EE, 0);
    // flash attention per (b, h, q-tile) -> f16
    attn_kernel<<<dim3(65, HH, BB), dim3(32), 0, stream>>>(Hbh, Obh);
    // y = o @ Wo^T + bo + x
    gemm_kernel<<<dim3(6, 130), blk, 0, stream>>>(Obh, woh + (size_t)i * EE * EE,
                                                  bo + i * EE, X, Y, nullptr, Mrows, EE, EE, 0);
    // x = LN1(y)
    layernorm_kernel<<<dim3(Mrows), dim3(128), 0, stream>>>(Y, (long long)EE,
                                                            ln1w + i * EE, ln1b + i * EE, X, Xh);
    // h = relu(x @ W1^T + b1)   (N=1536) -> f16 only
    gemm_kernel<<<dim3(24, 130), blk, 0, stream>>>(Xh, w1h + (size_t)i * FFD * EE,
                                                   b1 + i * FFD, nullptr, nullptr, Hbh,
                                                   Mrows, FFD, EE, 1);
    // y = h @ W2^T + b2 + x
    gemm_kernel<<<dim3(6, 130), blk, 0, stream>>>(Hbh, w2h + (size_t)i * EE * FFD,
                                                  b2 + i * EE, X, Y, nullptr, Mrows, EE, FFD, 0);
    // x = LN2(y)
    layernorm_kernel<<<dim3(Mrows), dim3(128), 0, stream>>>(Y, (long long)EE,
                                                            ln2w + i * EE, ln2b + i * EE, X, Xh);
  }

  // final LN on the 16 cls rows (rows b*TP of X)
  layernorm_kernel<<<dim3(BB), dim3(128), 0, stream>>>(X, (long long)TP * EE, lnfw, lnfb,
                                                       clsb, clsh);
  // logits = cls @ W_score^T + b_score   (M=16, N=1024, K=384)
  gemm_kernel<<<dim3(16, 1), blk, 0, stream>>>(clsh, wsch, bsc, nullptr, logits, nullptr,
                                               BB, 1024, EE, 0);
  // top-K=256 on (logits + gumbel)
  topk_kernel<<<dim3(BB), dim3(1024), 0, stream>>>(logits, gumbel, idxint, iout);
  // gather selected patches -> sub-patches + coords
  gather_kernel<<<dim3(KSEL, BB), dim3(256), 0, stream>>>(raw, coords, idxint, pout, cout);
}
